// DynamicEdgeGNNLayer_62843961475642
// MI455X (gfx1250) — matrix operand order repaired and checked by reading, not compiled, not run on hardware
//
#include <hip/hip_runtime.h>

#define B_  8
#define N_  512
#define H_  128
#define E_  32
#define HH_ 64
#define NEGV (-1e9f)
#define EPS_ 1e-5f

typedef _Float16 h16;
typedef __attribute__((ext_vector_type(16))) _Float16 v16h;
typedef __attribute__((ext_vector_type(8)))  _Float16 v8h;
typedef __attribute__((ext_vector_type(8)))  float    v8f;
typedef __attribute__((ext_vector_type(4)))  float    v4f;

static __device__ __forceinline__ v8f wmma16(v16h a, v16h b, v8f c) {
  // D = A(16x32 f16) * B(32x16 f16) + C(16x16 f32)
  return __builtin_amdgcn_wmma_f32_16x16x32_f16(false, a, false, b, (short)0, c, false, false);
}

// ---------------------------------------------------------------------------
// Kernel 1: edge MLP -> logits[B*N*N]. Dominant kernel (268 MB HBM stream).
// Each wave owns 8 consecutive 16-pair tiles (16 KB of edge_feats), processed
// two tiles per iteration: 8 independent WMMAs issued back-to-back, THEN the
// fused bias+ReLU+dot(W2) epilogue (avoids WMMA->VALU hazard v_nops).
// ---------------------------------------------------------------------------
__global__ void edge_logits_kernel(const float* __restrict__ edge,
                                   const float* __restrict__ W1,
                                   const float* __restrict__ b1,
                                   const float* __restrict__ W2,
                                   const float* __restrict__ b2,
                                   float* __restrict__ logits) {
  const int lane   = threadIdx.x & 31;
  const int waveId = (int)((blockIdx.x * blockDim.x + threadIdx.x) >> 5);
  const int half   = lane >> 4;   // 0: lanes 0-15, 1: lanes 16-31
  const int lm     = lane & 15;

  // ---- W1 B-fragments (held in registers for the whole kernel) ----
  // B 32x16 f16 layout: lane<16 -> K=0..7,16..23 of col lm ; lane>=16 -> K=8..15,24..31
  v16h bfrag[4];
  float b1v[4], w2v[4];
  const int kb0 = half * 16;
#pragma unroll
  for (int t = 0; t < 4; ++t) {
    const int col = t * 16 + lm;
#pragma unroll
    for (int r = 0; r < 16; ++r)
      bfrag[t][r] = (h16)W1[((r & 8) * 2 + half * 8 + (r & 7)) * HH_ + col];
    b1v[t] = b1[col];
    w2v[t] = W2[col];
  }
  const float b2v = b2[0];

  const int numTiles = (B_ * N_ * N_) / 16;   // 131072
  const int tilesPerWave = 8;                 // 131072 / 16384 waves
  const int t0 = waveId * tilesPerWave;
  if (t0 >= numTiles) return;

  const int e0 = half * 8;
  // A 16x32 f16 layout: lane<16 elems 0..7 = K0..7, elems 8..15 = K16..23
  //                     lane>=16 elems 0..7 = K8..15, elems 8..15 = K24..31
  for (int ii = 0; ii < tilesPerWave; ii += 2) {
    const long base = (long)(t0 + ii) * 16;
    const float* ep0 = edge + (size_t)(base + lm) * E_;
    const float* ep1 = ep0 + 16 * E_;          // next tile, +2 KB

    // prefetch next double-iteration's data for this lane (+4 KB)
    __builtin_prefetch(ep0 + 32 * E_, 0, 1);

    v16h a0, a1;
    {
      v4f x0 = *(const v4f*)(ep0 + e0);
      v4f x1 = *(const v4f*)(ep0 + e0 + 4);
      v4f x2 = *(const v4f*)(ep0 + e0 + 16);
      v4f x3 = *(const v4f*)(ep0 + e0 + 20);
      v4f y0 = *(const v4f*)(ep1 + e0);
      v4f y1 = *(const v4f*)(ep1 + e0 + 4);
      v4f y2 = *(const v4f*)(ep1 + e0 + 16);
      v4f y3 = *(const v4f*)(ep1 + e0 + 20);
#pragma unroll
      for (int i = 0; i < 4; ++i) {
        a0[i]      = (h16)x0[i];
        a0[4 + i]  = (h16)x1[i];
        a0[8 + i]  = (h16)x2[i];
        a0[12 + i] = (h16)x3[i];
        a1[i]      = (h16)y0[i];
        a1[4 + i]  = (h16)y1[i];
        a1[8 + i]  = (h16)y2[i];
        a1[12 + i] = (h16)y3[i];
      }
    }

    // ---- 8 independent WMMAs back-to-back ----
    v8f c0[4], c1[4];
#pragma unroll
    for (int t = 0; t < 4; ++t) {
      v8f z = {};
      c0[t] = wmma16(a0, bfrag[t], z);
      c1[t] = wmma16(a1, bfrag[t], z);
    }

    // ---- epilogue: bias + ReLU + * W2[col], partial over this lane's cols ----
    float p0[8] = {0,0,0,0,0,0,0,0};
    float p1[8] = {0,0,0,0,0,0,0,0};
#pragma unroll
    for (int t = 0; t < 4; ++t) {
#pragma unroll
      for (int r = 0; r < 8; ++r) {
        float h0 = c0[t][r] + b1v[t];
        float h1 = c1[t][r] + b1v[t];
        h0 = h0 > 0.f ? h0 : 0.f;
        h1 = h1 > 0.f ? h1 : 0.f;
        p0[r] += h0 * w2v[t];
        p1[r] += h1 * w2v[t];
      }
    }
    // Reduce across the 16 lanes of each half (cols) -> full row sums.
#pragma unroll
    for (int r = 0; r < 8; ++r) {
      p0[r] += __shfl_xor(p0[r], 1);  p1[r] += __shfl_xor(p1[r], 1);
      p0[r] += __shfl_xor(p0[r], 2);  p1[r] += __shfl_xor(p1[r], 2);
      p0[r] += __shfl_xor(p0[r], 4);  p1[r] += __shfl_xor(p1[r], 4);
      p0[r] += __shfl_xor(p0[r], 8);  p1[r] += __shfl_xor(p1[r], 8);
    }
    if (lm == 0) {   // lane 0 writes rows 0..7, lane 16 writes rows 8..15
      float* o0 = logits + base + half * 8;
      v4f s0 = {p0[0] + b2v, p0[1] + b2v, p0[2] + b2v, p0[3] + b2v};
      v4f s1 = {p0[4] + b2v, p0[5] + b2v, p0[6] + b2v, p0[7] + b2v};
      v4f s2 = {p1[0] + b2v, p1[1] + b2v, p1[2] + b2v, p1[3] + b2v};
      v4f s3 = {p1[4] + b2v, p1[5] + b2v, p1[6] + b2v, p1[7] + b2v};
      *(v4f*)(o0)          = s0;
      *(v4f*)(o0 + 4)      = s1;
      *(v4f*)(o0 + 16)     = s2;
      *(v4f*)(o0 + 16 + 4) = s3;
    }
  }
}

// ---------------------------------------------------------------------------
// Kernel 2: msgs = node_feats @ Wn + bn, stored TRANSPOSED as f16:
//           msT[(b*H + h) * N + j]  -> B-fragments in kernel 4 are contiguous.
// ---------------------------------------------------------------------------
__global__ void node_linear_kernel(const float* __restrict__ nodef,
                                   const float* __restrict__ Wn,
                                   const float* __restrict__ bn,
                                   h16* __restrict__ msT) {
  const int lane   = threadIdx.x & 31;
  const int waveId = (int)((blockIdx.x * blockDim.x + threadIdx.x) >> 5);
  const int nWaves = (int)((gridDim.x * blockDim.x) >> 5);
  const int half   = lane >> 4;
  const int lm     = lane & 15;

  const int numTiles = (B_ * N_) / 16;  // 256
  for (int tile = waveId; tile < numTiles; tile += nWaves) {
    const int rowBase = tile * 16;
    const float* np0 = nodef + (size_t)(rowBase + lm) * H_;

    v8f acc[8];
#pragma unroll
    for (int t = 0; t < 8; ++t) acc[t] = (v8f){};

#pragma unroll
    for (int ks = 0; ks < 4; ++ks) {         // K = 128 -> 4 steps of 32
      const int ka = ks * 32 + half * 8;
      v4f x0 = *(const v4f*)(np0 + ka);
      v4f x1 = *(const v4f*)(np0 + ka + 4);
      v4f x2 = *(const v4f*)(np0 + ka + 16);
      v4f x3 = *(const v4f*)(np0 + ka + 20);
      v16h a;
#pragma unroll
      for (int i = 0; i < 4; ++i) {
        a[i]      = (h16)x0[i];
        a[4 + i]  = (h16)x1[i];
        a[8 + i]  = (h16)x2[i];
        a[12 + i] = (h16)x3[i];
      }
      const int kb = ks * 32 + half * 8;
#pragma unroll
      for (int t = 0; t < 8; ++t) {
        const int col = t * 16 + lm;
        v16h bf;
#pragma unroll
        for (int r = 0; r < 16; ++r)
          bf[r] = (h16)Wn[(kb + (r & 8) * 2 + (r & 7)) * H_ + col];
        acc[t] = wmma16(a, bf, acc[t]);
      }
    }

    const int b = rowBase >> 9;              // tiles never cross a batch
#pragma unroll
    for (int t = 0; t < 8; ++t) {
      const int col = t * 16 + lm;
      const float bnv = bn[col];
#pragma unroll
      for (int r = 0; r < 8; ++r) {
        const int m = rowBase + half * 8 + r;        // flat row b*N + j
        const int j = m & (N_ - 1);
        msT[(size_t)(b * H_ + col) * N_ + j] = (h16)(acc[t][r] + bnv);
      }
    }
  }
}

// ---------------------------------------------------------------------------
// Kernel 3: masked row softmax -> attention stored as f16 row-major.
// One block (256 thr) per (b,i) row of 512 scores.
// ---------------------------------------------------------------------------
__global__ void softmax_kernel(const float* __restrict__ logits,
                               const float* __restrict__ mask,
                               h16* __restrict__ attn) {
  __shared__ float redmax[8];
  __shared__ float redsum[8];
  const int row = blockIdx.x;             // b*N + i
  const int b   = row >> 9;
  const int i   = row & (N_ - 1);
  const int tid = threadIdx.x;
  const float mi = mask[b * N_ + i];
  const float* lrow = logits + (size_t)row * N_;

  float s0, s1;
  {
    const int j0 = tid, j1 = tid + 256;
    const float m0 = mask[b * N_ + j0];
    const float m1 = mask[b * N_ + j1];
    s0 = (mi * m0 != 0.f && j0 != i) ? lrow[j0] : NEGV;
    s1 = (mi * m1 != 0.f && j1 != i) ? lrow[j1] : NEGV;
  }
  float mymax = fmaxf(s0, s1);
#pragma unroll
  for (int m = 16; m > 0; m >>= 1) mymax = fmaxf(mymax, __shfl_xor(mymax, m));
  if ((tid & 31) == 0) redmax[tid >> 5] = mymax;
  __syncthreads();
  float bmax = redmax[0];
#pragma unroll
  for (int w = 1; w < 8; ++w) bmax = fmaxf(bmax, redmax[w]);

  const float e0 = __expf(s0 - bmax);
  const float e1 = __expf(s1 - bmax);
  float mysum = e0 + e1;
#pragma unroll
  for (int m = 16; m > 0; m >>= 1) mysum += __shfl_xor(mysum, m);
  if ((tid & 31) == 0) redsum[tid >> 5] = mysum;
  __syncthreads();
  float bsum = 0.f;
#pragma unroll
  for (int w = 0; w < 8; ++w) bsum += redsum[w];
  const float inv = 1.f / bsum;

  h16* arow = attn + (size_t)row * N_;
  arow[tid]       = (h16)(e0 * inv);
  arow[tid + 256] = (h16)(e1 * inv);
}

// ---------------------------------------------------------------------------
// Kernel 4: updated = attn @ msgs (WMMA, K=512), + residual, + LayerNorm.
// One wave per 16-row i-tile; 256 waves total.
// ---------------------------------------------------------------------------
__global__ void update_ln_kernel(const h16* __restrict__ attn,
                                 const h16* __restrict__ msT,
                                 const float* __restrict__ nodef,
                                 const float* __restrict__ gamma,
                                 const float* __restrict__ beta,
                                 float* __restrict__ out) {
  const int lane   = threadIdx.x & 31;
  const int waveId = (int)((blockIdx.x * blockDim.x + threadIdx.x) >> 5);
  const int half   = lane >> 4;
  const int lm     = lane & 15;

  const int rowBase = waveId * 16;        // flat b*N + i base
  const int b = rowBase >> 9;

  v8f acc[8];
#pragma unroll
  for (int t = 0; t < 8; ++t) acc[t] = (v8f){};

  const h16* arow = attn + (size_t)(rowBase + lm) * N_;
  for (int ks = 0; ks < 16; ++ks) {       // K = 512 -> 16 steps of 32
    const h16* ap = arow + ks * 32 + half * 8;
    v8h alo = *(const v8h*)(ap);
    v8h ahi = *(const v8h*)(ap + 16);
    v16h a;
#pragma unroll
    for (int i = 0; i < 8; ++i) { a[i] = alo[i]; a[8 + i] = ahi[i]; }

    const int kb = ks * 32 + half * 8;
#pragma unroll
    for (int t = 0; t < 8; ++t) {
      const int col = t * 16 + lm;
      const h16* bp = msT + (size_t)(b * H_ + col) * N_ + kb;
      v8h blo = *(const v8h*)(bp);
      v8h bhi = *(const v8h*)(bp + 16);
      v16h bf;
#pragma unroll
      for (int i = 0; i < 8; ++i) { bf[i] = blo[i]; bf[8 + i] = bhi[i]; }
      acc[t] = wmma16(a, bf, acc[t]);
    }
  }

  // x = node_feats + updated ; accumulate row mean / mean-of-squares
  float s1[8] = {0,0,0,0,0,0,0,0};
  float s2[8] = {0,0,0,0,0,0,0,0};
#pragma unroll
  for (int t = 0; t < 8; ++t) {
    const int col = t * 16 + lm;
#pragma unroll
    for (int r = 0; r < 8; ++r) {
      const int m = rowBase + half * 8 + r;
      const float x = acc[t][r] + nodef[(size_t)m * H_ + col];
      acc[t][r] = x;
      s1[r] += x;
      s2[r] += x * x;
    }
  }
#pragma unroll
  for (int r = 0; r < 8; ++r) {
    s1[r] += __shfl_xor(s1[r], 1); s2[r] += __shfl_xor(s2[r], 1);
    s1[r] += __shfl_xor(s1[r], 2); s2[r] += __shfl_xor(s2[r], 2);
    s1[r] += __shfl_xor(s1[r], 4); s2[r] += __shfl_xor(s2[r], 4);
    s1[r] += __shfl_xor(s1[r], 8); s2[r] += __shfl_xor(s2[r], 8);
  }
  float mean[8], rstd[8];
#pragma unroll
  for (int r = 0; r < 8; ++r) {
    mean[r] = s1[r] * (1.0f / H_);
    const float var = s2[r] * (1.0f / H_) - mean[r] * mean[r];
    rstd[r] = rsqrtf(var + EPS_);
  }
#pragma unroll
  for (int t = 0; t < 8; ++t) {
    const int col = t * 16 + lm;
    const float g = gamma[col], be = beta[col];
#pragma unroll
    for (int r = 0; r < 8; ++r) {
      const int m = rowBase + half * 8 + r;
      out[(size_t)m * H_ + col] = (acc[t][r] - mean[r]) * rstd[r] * g + be;
    }
  }
}

// ---------------------------------------------------------------------------
extern "C" void kernel_launch(void* const* d_in, const int* in_sizes, int n_in,
                              void* d_out, int out_size, void* d_ws, size_t ws_size,
                              hipStream_t stream) {
  (void)in_sizes; (void)n_in; (void)out_size; (void)ws_size;
  const float* node_feats = (const float*)d_in[0];
  const float* edge_feats = (const float*)d_in[1];
  const float* node_mask  = (const float*)d_in[2];
  const float* W1 = (const float*)d_in[3];
  const float* b1 = (const float*)d_in[4];
  const float* W2 = (const float*)d_in[5];
  const float* b2 = (const float*)d_in[6];
  const float* Wn = (const float*)d_in[7];
  const float* bn = (const float*)d_in[8];
  const float* gamma = (const float*)d_in[9];
  const float* beta  = (const float*)d_in[10];
  float* out = (float*)d_out;

  char* ws = (char*)d_ws;
  float* logits = (float*)ws;                                  // 8 MB  (B*N*N f32)
  h16*   msT    = (h16*)(ws + (size_t)8 * 1024 * 1024);        // 1 MB  (B*H*N f16, transposed)
  h16*   attnp  = (h16*)(ws + (size_t)9 * 1024 * 1024);        // 4 MB  (B*N*N f16)

  hipLaunchKernelGGL(edge_logits_kernel, dim3(2048), dim3(256), 0, stream,
                     edge_feats, W1, b1, W2, b2, logits);
  hipLaunchKernelGGL(node_linear_kernel, dim3(32), dim3(256), 0, stream,
                     node_feats, Wn, bn, msT);
  hipLaunchKernelGGL(softmax_kernel, dim3(B_ * N_), dim3(256), 0, stream,
                     logits, node_mask, attnp);
  hipLaunchKernelGGL(update_ln_kernel, dim3(32), dim3(256), 0, stream,
                     attnp, msT, node_feats, gamma, beta, out);
}
